// NonLocalBlock3D_1185410973938
// MI455X (gfx1250) — matrix operand order repaired and checked by reading, not compiled, not run on hardware
//
#include <hip/hip_runtime.h>
#include <hip/hip_bf16.h>
#include <stdint.h>

// ---------------------------------------------------------------------------
// NonLocalBlock3D on MI455X (gfx1250): bf16 WMMA pipeline.
//   B=2, C=256, Ci=128, N=65536.
//   proj[b][0:128]=theta, [128:256]=phi, [256:384]=g   (bf16)
//   kv = ph @ g^T  (f32, split-K partials reduced deterministically)
//   M  = W_w @ kv^T / N  (bf16, [256][128])
//   out = x + W_b + M @ th
// LDS tiles padded to 20-word row stride: conflict-free ds_load_b128 /
// transpose stores on the 64x4B banked LDS, 16B alignment preserved.
// ---------------------------------------------------------------------------

typedef __attribute__((ext_vector_type(16))) __bf16 v16bf;
typedef __attribute__((ext_vector_type(8)))  float  v8f;

#define LDA 40   // A-tile row stride in bf16 (80 B)
#define LDB 20   // B-tile row stride in u32  (80 B)

union FragAB {
    uint4 q[2];
    v16bf v;
};

// ---------------- bf16 conversion: v_cvt_pk_bf16_f32 ------------------------
__device__ __forceinline__ uint16_t f2bf_sw(float f) {
    union { float f; uint32_t u; } x;
    x.f = f;
    uint32_t r = x.u + 0x7FFFu + ((x.u >> 16) & 1u);   // round-to-nearest-even
    return (uint16_t)(r >> 16);
}

__device__ __forceinline__ uint32_t pack2_bf16(float lo, float hi) {
#if defined(__AMDGCN__)
    uint32_t r;
    asm("v_cvt_pk_bf16_f32 %0, %1, %2" : "=v"(r) : "v"(lo), "v"(hi));
    return r;
#else
    return (uint32_t)f2bf_sw(lo) | ((uint32_t)f2bf_sw(hi) << 16);
#endif
}

__device__ __forceinline__ uint16_t f2bf(float f) {
    return (uint16_t)pack2_bf16(f, 0.0f);
}

// ---------------- CDNA5 async global->LDS copy (ASYNCcnt) -------------------
#if defined(__AMDGCN__) && __has_builtin(__builtin_amdgcn_global_load_async_to_lds_b128)
#define HAVE_ASYNC_LDS 1
#else
#define HAVE_ASYNC_LDS 0
#endif

typedef int gv4i __attribute__((vector_size(16)));   // matches builtin prototype

__device__ __forceinline__ void async_copy_b128(void* ldst, const void* gsrc) {
#if HAVE_ASYNC_LDS
    auto g = (__attribute__((address_space(1))) gv4i*)(uintptr_t)gsrc;
    auto l = (__attribute__((address_space(3))) gv4i*)(uint32_t)(uintptr_t)ldst;
    __builtin_amdgcn_global_load_async_to_lds_b128(g, l, 0, 0);
#else
    *(uint4*)ldst = *(const uint4*)gsrc;
#endif
}

#if HAVE_ASYNC_LDS
#if __has_builtin(__builtin_amdgcn_s_wait_asynccnt)
#define ASYNC_WAIT(n) __builtin_amdgcn_s_wait_asynccnt(n)
#else
#define ASYNC_WAIT(n) asm volatile("s_wait_asynccnt %0" ::"i"(n) : "memory")
#endif
#else
#define ASYNC_WAIT(n) do { } while (0)
#endif

__device__ __forceinline__ v8f v8f_zero() {
    v8f z;
#pragma unroll
    for (int e = 0; e < 8; ++e) z[e] = 0.0f;
    return z;
}

// Load A fragment (16x32 bf16) from LDS tile row (LDA-strided, 32 valid bf16).
// Lane l: m = base + (l&15); hi = l>>4 selects K halves per ISA layout:
//   VGPR0-3 : K = 8*hi .. 8*hi+7 ; VGPR4-7 : K = 16+8*hi .. +7
__device__ __forceinline__ FragAB load_fragA(const uint16_t* row, int hi) {
    FragAB f;
    const uint4* rp = (const uint4*)row;
    f.q[0] = rp[hi];
    f.q[1] = rp[2 + hi];
    return f;
}

// Load B fragment (32x16 bf16) from LDS tile row stored as 16 valid u32 k-pairs.
// Lane l: n = base + (l&15); hi = l>>4: K = 8*hi..8*hi+7, 16+8*hi..16+8*hi+7 (pair p holds K=2p,2p+1)
__device__ __forceinline__ FragAB load_fragB(const uint32_t* row, int hi) {
    FragAB f;
    const uint4* rp = (const uint4*)row;
    f.q[0] = rp[hi];
    f.q[1] = rp[2 + hi];
    return f;
}

// ---------------------------------------------------------------------------
// 1) Pack projection weights: Wc bf16 [384][256], biasc f32 [384]
// ---------------------------------------------------------------------------
__global__ void pack_weights(const float* __restrict__ g_w, const float* __restrict__ g_b,
                             const float* __restrict__ t_w, const float* __restrict__ t_b,
                             const float* __restrict__ p_w, const float* __restrict__ p_b,
                             uint16_t* __restrict__ Wc, float* __restrict__ biasc) {
    int r = blockIdx.x;     // 0..383
    int c = threadIdx.x;    // 0..255
    const float* src; const float* bsrc; int rr;
    if (r < 128)      { src = t_w; bsrc = t_b; rr = r; }
    else if (r < 256) { src = p_w; bsrc = p_b; rr = r - 128; }
    else              { src = g_w; bsrc = g_b; rr = r - 256; }
    Wc[r * 256 + c] = f2bf(src[rr * 256 + c]);
    if (c == 0) biasc[r] = bsrc[rr];
}

// ---------------------------------------------------------------------------
// 2) Projection GEMM: proj[b][m][n] = Wc[m][:] . x[b][:][n] + biasc[m]
//    grid (N/128, 3, B), block 256 (8 waves, 4(M)x2(N) wave grid)
// ---------------------------------------------------------------------------
__global__ __launch_bounds__(256) void proj_gemm(const float* __restrict__ x,
                                                 const uint16_t* __restrict__ Wc,
                                                 const float* __restrict__ biasc,
                                                 uint16_t* __restrict__ proj) {
    __shared__ uint16_t ldsA[128][LDA];   // A tile: rows m, cols k (bf16)
    __shared__ uint32_t ldsB[128][LDB];   // B tile transposed: [n][kpair]

    const int n0 = blockIdx.x * 128;
    const int m0 = blockIdx.y * 128;
    const int b  = blockIdx.z;
    const int t = threadIdx.x;
    const int wave = t >> 5, lane = t & 31;
    const int wm = wave >> 1, wn = wave & 1;
    const int lr = lane & 15, hi = lane >> 4;

    v8f acc[2][4];
#pragma unroll
    for (int i = 0; i < 2; ++i)
#pragma unroll
        for (int j = 0; j < 4; ++j) acc[i][j] = v8f_zero();

    for (int ks = 0; ks < 256; ks += 32) {
        // A tile (straight bf16 copy) via async global->LDS
        {
            int row = t >> 1, hf = t & 1;
            const uint16_t* src = Wc + (size_t)(m0 + row) * 256 + ks + hf * 16;
            async_copy_b128(&ldsA[row][hf * 16 + 0], src + 0);
            async_copy_b128(&ldsA[row][hf * 16 + 8], src + 8);
        }
        // B tile: x[b][ks+r][n0+c] f32 -> bf16 pairs, transposed into ldsB[n][kpair]
        {
            int r0 = (t >> 4) * 2;      // 0..30 step 2
            int c0 = (t & 15) * 8;      // 0..120 step 8
            const float* pa = x + (((size_t)b * 256 + ks + r0) * 65536) + n0 + c0;
            const float* pb = pa + 65536;
            float4 a0 = ((const float4*)pa)[0], a1 = ((const float4*)pa)[1];
            float4 b0 = ((const float4*)pb)[0], b1 = ((const float4*)pb)[1];
            if (ks + 32 < 256) __builtin_prefetch(pa + 2u * 65536u, 0, 0);
            float av[8] = {a0.x, a0.y, a0.z, a0.w, a1.x, a1.y, a1.z, a1.w};
            float bv[8] = {b0.x, b0.y, b0.z, b0.w, b1.x, b1.y, b1.z, b1.w};
#pragma unroll
            for (int j = 0; j < 8; ++j)
                ldsB[c0 + j][r0 >> 1] = pack2_bf16(av[j], bv[j]);
        }
        ASYNC_WAIT(0);
        __syncthreads();

        FragAB afr[2], bfr[4];
#pragma unroll
        for (int ms = 0; ms < 2; ++ms)
            afr[ms] = load_fragA(&ldsA[wm * 32 + ms * 16 + lr][0], hi);
#pragma unroll
        for (int ns = 0; ns < 4; ++ns)
            bfr[ns] = load_fragB(&ldsB[wn * 64 + ns * 16 + lr][0], hi);
#pragma unroll
        for (int ms = 0; ms < 2; ++ms)
#pragma unroll
            for (int ns = 0; ns < 4; ++ns)
                acc[ms][ns] = __builtin_amdgcn_wmma_f32_16x16x32_bf16(
                    false, afr[ms].v, false, bfr[ns].v, (short)0, acc[ms][ns], false, false);
        __syncthreads();
    }

    // epilogue: + bias, store bf16
#pragma unroll
    for (int ms = 0; ms < 2; ++ms) {
#pragma unroll
        for (int ns = 0; ns < 4; ++ns) {
            int bm = m0 + wm * 32 + ms * 16;
            int bn = n0 + wn * 64 + ns * 16 + lr;
#pragma unroll
            for (int v = 0; v < 8; ++v) {
                int m = bm + v + 8 * hi;
                float val = acc[ms][ns][v] + biasc[m];
                proj[((size_t)b * 384 + m) * 65536 + bn] = f2bf(val);
            }
        }
    }
}

// ---------------------------------------------------------------------------
// 3) kv partials: kvpart[b][chunk][i][j] = sum_{k in chunk(1024)} ph[i][k]*g[j][k]
//    grid (64, B), block 256.  Double-buffered async global->LDS pipeline.
// ---------------------------------------------------------------------------
__global__ __launch_bounds__(256) void kv_gemm(const uint16_t* __restrict__ proj,
                                               float* __restrict__ kvpart) {
    __shared__ uint16_t ldsA[2][128][LDA];
    __shared__ uint32_t ldsB[2][128][LDB];

    const int b  = blockIdx.y;
    const int k0 = blockIdx.x * 1024;
    const int t = threadIdx.x;
    const int wave = t >> 5, lane = t & 31;
    const int wm = wave >> 1, wn = wave & 1;
    const int lr = lane & 15, hi = lane >> 4;
    const int row = t >> 1, hf = t & 1;

    const uint16_t* ph = proj + ((size_t)b * 384 + 128) * 65536;
    const uint16_t* gg = proj + ((size_t)b * 384 + 256) * 65536;

    v8f acc[2][4];
#pragma unroll
    for (int i = 0; i < 2; ++i)
#pragma unroll
        for (int j = 0; j < 4; ++j) acc[i][j] = v8f_zero();

    // Issue one K-tile (4 x b128 per thread) into LDS buffer `buf`.
    auto issue = [&](int buf, int ks) {
        const uint16_t* sa = ph + (size_t)row * 65536 + ks + hf * 16;   // A: ph[i][k]
        async_copy_b128(&ldsA[buf][row][hf * 16 + 0], sa + 0);
        async_copy_b128(&ldsA[buf][row][hf * 16 + 8], sa + 8);
        const uint16_t* sb = gg + (size_t)row * 65536 + ks + hf * 16;   // B[k][j]=g[j][k]
        async_copy_b128(&ldsB[buf][row][hf * 8 + 0], sb + 0);
        async_copy_b128(&ldsB[buf][row][hf * 8 + 4], sb + 8);
    };

    issue(0, k0);
    int cur = 0;
    for (int i = 0; i < 32; ++i) {
        if (i + 1 < 32) {
            issue(cur ^ 1, k0 + (i + 1) * 32);
            ASYNC_WAIT(4);            // tile i complete; tile i+1 still in flight
        } else {
            ASYNC_WAIT(0);
        }
        __syncthreads();

        FragAB afr[2], bfr[4];
#pragma unroll
        for (int ms = 0; ms < 2; ++ms)
            afr[ms] = load_fragA(&ldsA[cur][wm * 32 + ms * 16 + lr][0], hi);
#pragma unroll
        for (int ns = 0; ns < 4; ++ns)
            bfr[ns] = load_fragB(&ldsB[cur][wn * 64 + ns * 16 + lr][0], hi);
#pragma unroll
        for (int ms = 0; ms < 2; ++ms)
#pragma unroll
            for (int ns = 0; ns < 4; ++ns)
                acc[ms][ns] = __builtin_amdgcn_wmma_f32_16x16x32_bf16(
                    false, afr[ms].v, false, bfr[ns].v, (short)0, acc[ms][ns], false, false);
        __syncthreads();
        cur ^= 1;
    }

    float* dst = kvpart + (((size_t)b * 64 + blockIdx.x) * 16384);
#pragma unroll
    for (int ms = 0; ms < 2; ++ms)
#pragma unroll
        for (int ns = 0; ns < 4; ++ns)
#pragma unroll
            for (int v = 0; v < 8; ++v) {
                int i = wm * 32 + ms * 16 + v + 8 * hi;   // ph row
                int j = wn * 64 + ns * 16 + lr;           // g row
                dst[i * 128 + j] = acc[ms][ns][v];
            }
}

// Deterministic reduction of split-K partials. grid (128, B), block 128.
__global__ void kv_reduce(const float* __restrict__ part, float* __restrict__ kv) {
    int i = blockIdx.x, b = blockIdx.y, j = threadIdx.x;
    const float* p = part + ((size_t)b * 64) * 16384 + i * 128 + j;
    float s = 0.0f;
    for (int c = 0; c < 64; ++c) s += p[(size_t)c * 16384];
    kv[((size_t)b * 128 + i) * 128 + j] = s;
}

// ---------------------------------------------------------------------------
// 4) Fold output conv into kv: M[b][o][i] = (1/N) sum_j W_w[o][j]*kv[b][i][j]
//    grid (256, B), block 128
// ---------------------------------------------------------------------------
__global__ void mmat_kernel(const float* __restrict__ Ww, const float* __restrict__ kv,
                            uint16_t* __restrict__ Mmat) {
    int o = blockIdx.x, b = blockIdx.y, i = threadIdx.x;
    const float* wr = Ww + (size_t)o * 128;
    const float* kr = kv + ((size_t)b * 128 + i) * 128;
    float s = 0.0f;
    for (int j = 0; j < 128; ++j) s += wr[j] * kr[j];
    Mmat[((size_t)b * 256 + o) * 128 + i] = f2bf(s * (1.0f / 65536.0f));
}

// ---------------------------------------------------------------------------
// 5) Output GEMM + bias + residual:
//    out[b][m][n] = x[b][m][n] + W_b[m] + sum_i Mmat[b][m][i]*th[i][n]
//    grid (N/128, 2, B), block 256
// ---------------------------------------------------------------------------
__global__ __launch_bounds__(256) void final_gemm(const uint16_t* __restrict__ proj,
                                                  const uint16_t* __restrict__ Mmat,
                                                  const float* __restrict__ Wb,
                                                  const float* __restrict__ x,
                                                  float* __restrict__ out) {
    __shared__ uint16_t ldsA[128][LDA];
    __shared__ uint32_t ldsB[128][LDB];

    const int n0 = blockIdx.x * 128;
    const int m0 = blockIdx.y * 128;
    const int b  = blockIdx.z;
    const int t = threadIdx.x;
    const int wave = t >> 5, lane = t & 31;
    const int wm = wave >> 1, wn = wave & 1;
    const int lr = lane & 15, hi = lane >> 4;

    const uint16_t* th = proj + (size_t)b * 384 * 65536;        // rows 0..127
    const uint16_t* A  = Mmat + (size_t)b * 256 * 128;

    v8f acc[2][4];
#pragma unroll
    for (int i = 0; i < 2; ++i)
#pragma unroll
        for (int j = 0; j < 4; ++j) acc[i][j] = v8f_zero();

    for (int ks = 0; ks < 128; ks += 32) {
        {   // A tile (straight bf16 copy) via async global->LDS
            int row = t >> 1, hf = t & 1;
            const uint16_t* src = A + (size_t)(m0 + row) * 128 + ks + hf * 16;
            async_copy_b128(&ldsA[row][hf * 16 + 0], src + 0);
            async_copy_b128(&ldsA[row][hf * 16 + 8], src + 8);
        }
        {   // B tile: th[ks+r][n0+c] bf16 -> transposed pairs into ldsB[n][kpair]
            int r0 = (t >> 4) * 2;
            int c0 = (t & 15) * 8;
            const uint16_t* pa = th + (size_t)(ks + r0) * 65536 + n0 + c0;
            const uint16_t* pb = pa + 65536;
            union { uint4 q; uint16_t s[8]; } ua, ub;
            ua.q = *(const uint4*)pa;
            ub.q = *(const uint4*)pb;
#pragma unroll
            for (int j = 0; j < 8; ++j)
                ldsB[c0 + j][r0 >> 1] = (uint32_t)ua.s[j] | ((uint32_t)ub.s[j] << 16);
        }
        ASYNC_WAIT(0);
        __syncthreads();

        FragAB afr[2], bfr[4];
#pragma unroll
        for (int ms = 0; ms < 2; ++ms)
            afr[ms] = load_fragA(&ldsA[wm * 32 + ms * 16 + lr][0], hi);
#pragma unroll
        for (int ns = 0; ns < 4; ++ns)
            bfr[ns] = load_fragB(&ldsB[wn * 64 + ns * 16 + lr][0], hi);
#pragma unroll
        for (int ms = 0; ms < 2; ++ms)
#pragma unroll
            for (int ns = 0; ns < 4; ++ns)
                acc[ms][ns] = __builtin_amdgcn_wmma_f32_16x16x32_bf16(
                    false, afr[ms].v, false, bfr[ns].v, (short)0, acc[ms][ns], false, false);
        __syncthreads();
    }

    // epilogue: + W_b + residual x, store f32
#pragma unroll
    for (int ms = 0; ms < 2; ++ms) {
#pragma unroll
        for (int ns = 0; ns < 4; ++ns) {
            int bm = m0 + wm * 32 + ms * 16;
            int bn = n0 + wn * 64 + ns * 16 + lr;
#pragma unroll
            for (int v = 0; v < 8; ++v) {
                int m = bm + v + 8 * hi;
                size_t idx = ((size_t)b * 256 + m) * 65536 + bn;
                out[idx] = acc[ms][ns][v] + Wb[m] + x[idx];
            }
        }
    }
}

// ---------------------------------------------------------------------------
// Host launch
// ---------------------------------------------------------------------------
extern "C" void kernel_launch(void* const* d_in, const int* in_sizes, int n_in,
                              void* d_out, int out_size, void* d_ws, size_t ws_size,
                              hipStream_t stream) {
    const float* x       = (const float*)d_in[0];
    const float* g_w     = (const float*)d_in[1];
    const float* g_b     = (const float*)d_in[2];
    const float* theta_w = (const float*)d_in[3];
    const float* theta_b = (const float*)d_in[4];
    const float* phi_w   = (const float*)d_in[5];
    const float* phi_b   = (const float*)d_in[6];
    const float* W_w     = (const float*)d_in[7];
    const float* W_b     = (const float*)d_in[8];
    float* out = (float*)d_out;

    char* ws = (char*)d_ws;
    size_t off = 0;
    uint16_t* proj = (uint16_t*)(ws + off); off += (size_t)2 * 384 * 65536 * 2;  // 96 MiB bf16
    float* kvpart  = (float*)(ws + off);    off += (size_t)2 * 64 * 16384 * 4;   // 8 MiB
    float* kv      = (float*)(ws + off);    off += (size_t)2 * 128 * 128 * 4;
    uint16_t* Mmat = (uint16_t*)(ws + off); off += (size_t)2 * 256 * 128 * 2;
    uint16_t* Wc   = (uint16_t*)(ws + off); off += (size_t)384 * 256 * 2;
    float* biasc   = (float*)(ws + off);    off += (size_t)384 * 4;
    (void)ws_size; (void)in_sizes; (void)n_in; (void)out_size;

    pack_weights<<<dim3(384), dim3(256), 0, stream>>>(g_w, g_b, theta_w, theta_b,
                                                      phi_w, phi_b, Wc, biasc);
    proj_gemm<<<dim3(512, 3, 2), dim3(256), 0, stream>>>(x, Wc, biasc, proj);
    kv_gemm<<<dim3(64, 2), dim3(256), 0, stream>>>(proj, kvpart);
    kv_reduce<<<dim3(128, 2), dim3(128), 0, stream>>>(kvpart, kv);
    mmat_kernel<<<dim3(256, 2), dim3(128), 0, stream>>>(W_w, kv, Mmat);
    final_gemm<<<dim3(512, 2, 2), dim3(256), 0, stream>>>(proj, Mmat, W_b, x, out);
}
